// MultiHeadAttention_59279138620011
// MI455X (gfx1250) — matrix operand order repaired and checked by reading, not compiled, not run on hardware
//
#include <hip/hip_runtime.h>
#include <hip/hip_bf16.h>

// MHA forward for B=4,T=2048,C=1024,H=16,HS=64 on gfx1250 (wave32, WMMA).
// (0) fp32->bf16 convert of x / transposed W, (1) QKV projection via
// v_wmma_f32_16x16x32_bf16, (2) flash attention with workgroup-cooperative
// async K/V staging into LDS (triple-buffered, 1 barrier/tile).

typedef __attribute__((ext_vector_type(16))) __bf16 v16bf;
typedef __attribute__((ext_vector_type(8)))  __bf16 v8bf;
typedef __attribute__((ext_vector_type(4)))  __bf16 v4bf;
typedef __attribute__((ext_vector_type(8)))  float  v8f;
typedef __attribute__((ext_vector_type(4)))  int    v4i;

#define BB 4
#define TT 2048
#define CC 1024
#define HH 16
#define DD 64

#define ASPACE(n) __attribute__((address_space(n)))

#if __has_builtin(__builtin_amdgcn_global_load_async_to_lds_b128)
#define HAVE_ASYNC_LDS 1
#else
#define HAVE_ASYNC_LDS 0
#endif

__device__ __forceinline__ v8f wmma_bf16(v16bf a, v16bf b, v8f c) {
  return __builtin_amdgcn_wmma_f32_16x16x32_bf16(false, a, false, b, (short)0, c,
                                                 false, false);
}

__device__ __forceinline__ v16bf cat8(v8bf lo, v8bf hi) {
  return __builtin_shufflevector(lo, hi, 0, 1, 2, 3, 4, 5, 6, 7, 8, 9, 10, 11,
                                 12, 13, 14, 15);
}

__device__ __forceinline__ v8f zero8() {
  v8f z;
#pragma unroll
  for (int i = 0; i < 8; ++i) z[i] = 0.0f;
  return z;
}

// 16-byte global -> LDS stage; async DMA (ASYNCcnt) when available.
// Builtin signature (from hipcc diagnostic): (v4i as1*, v4i as3*, imm, imm).
__device__ __forceinline__ void stage16(const __bf16* g, __bf16* l) {
#if HAVE_ASYNC_LDS
  ASPACE(1) v4i* gp = (ASPACE(1) v4i*)(ASPACE(1) void*)(void*)g;
  ASPACE(3) v4i* lp = (ASPACE(3) v4i*)(ASPACE(3) void*)l;
  __builtin_amdgcn_global_load_async_to_lds_b128(gp, lp, 0, 0);
#else
  *(v8bf*)l = *(const v8bf*)g;  // fallback: load + ds_store
#endif
}

// A-fragment (16 rows x 32 contraction, bf16), ISA 7.12.2 layout:
// lane<16: row=lane, K=k0+0..7,k0+16..23; lane>=16: row=lane-16, K=+8..15,+24..31
__device__ __forceinline__ v16bf load_a(const __bf16* __restrict__ base,
                                        int row_stride, int k0, int lane) {
  const int row = lane & 15;
  const int koff = (lane < 16) ? 0 : 8;
  const __bf16* p = base + (size_t)row * row_stride + k0 + koff;
  v8bf lo = *(const v8bf*)p;
  v8bf hi = *(const v8bf*)(p + 16);
  return cat8(lo, hi);
}

// B-fragment (32 contraction x 16 cols): contraction contiguous per column.
// lane<16 -> col=lane, K=k0+0..7,k0+16..23; lane>=16 -> col=lane-16, K=+8..15,+24..31.
__device__ __forceinline__ v16bf load_bt(const __bf16* colbase,
                                         size_t col_stride, int k0, int lane) {
  const int col = lane & 15;
  const int kb = k0 + ((lane < 16) ? 0 : 8);
  const __bf16* p = colbase + (size_t)col * col_stride + kb;
  return cat8(*(const v8bf*)p, *(const v8bf*)(p + 16));
}

// ---------------- Kernel 0a: x fp32 -> bf16 ----------------
__global__ void __launch_bounds__(256) mha_cvt_x(const float* __restrict__ in,
                                                 __bf16* __restrict__ out, int n) {
  int i = (blockIdx.x * 256 + threadIdx.x) * 4;
  if (i >= n) return;
  float4 f = *(const float4*)(in + i);
  v4bf o;
  o[0] = (__bf16)f.x; o[1] = (__bf16)f.y; o[2] = (__bf16)f.z; o[3] = (__bf16)f.w;
  *(v4bf*)(out + i) = o;
}

// ------- Kernel 0b: W[H,C,D] fp32 -> Wt[H,D,C] bf16 (optionally scaled) -------
__global__ void __launch_bounds__(256) mha_cvt_w(const float* __restrict__ in,
                                                 __bf16* __restrict__ out,
                                                 float scale) {
  int i = blockIdx.x * 256 + threadIdx.x;
  int h = i / (CC * DD);
  int r = i % (CC * DD);
  int c = r / DD;
  int d = r % DD;
  out[(size_t)h * DD * CC + (size_t)d * CC + c] = (__bf16)(in[i] * scale);
}

// ---------------- Kernel 1: QKV projection ----------------
// grid (T/128, B*H), 8 waves; wave w computes rows t0..t0+15 of q,k,v.
__global__ void __launch_bounds__(256) mha_proj(
    const __bf16* __restrict__ xb, const __bf16* __restrict__ wqt,
    const __bf16* __restrict__ wkt, const __bf16* __restrict__ wvt,
    __bf16* __restrict__ q, __bf16* __restrict__ k, __bf16* __restrict__ vt) {
  const int lane = threadIdx.x & 31;
  const int wave = threadIdx.x >> 5;
  const int t0 = blockIdx.x * 128 + wave * 16;
  const int b = blockIdx.y / HH;
  const int h = blockIdx.y % HH;

  const __bf16* xrow = xb + ((size_t)b * TT + t0) * CC;
  const __bf16* wq = wqt + (size_t)h * DD * CC;
  const __bf16* wk = wkt + (size_t)h * DD * CC;
  const __bf16* wv = wvt + (size_t)h * DD * CC;

  v8f aq[4], ak[4], av[4];
#pragma unroll
  for (int j = 0; j < 4; ++j) { aq[j] = zero8(); ak[j] = zero8(); av[j] = zero8(); }

  for (int c0 = 0; c0 < CC; c0 += 32) {
    v16bf a = load_a(xrow, CC, c0, lane);  // one x fragment feeds 12 WMMAs
#pragma unroll
    for (int j = 0; j < 4; ++j) {
      v16bf bq = load_bt(wq + (size_t)(j * 16) * CC, CC, c0, lane);
      v16bf bk = load_bt(wk + (size_t)(j * 16) * CC, CC, c0, lane);
      v16bf bv = load_bt(wv + (size_t)(j * 16) * CC, CC, c0, lane);
      aq[j] = wmma_bf16(a, bq, aq[j]);
      ak[j] = wmma_bf16(a, bk, ak[j]);
      av[j] = wmma_bf16(a, bv, av[j]);
    }
  }

  const int n = lane & 15;
  const int hi = lane >> 4;
  const size_t qkbase = (size_t)(b * HH + h) * TT;
#pragma unroll
  for (int j = 0; j < 4; ++j) {
#pragma unroll
    for (int r = 0; r < 8; ++r) {
      int t = t0 + r + hi * 8;
      q[(qkbase + t) * DD + j * 16 + n] = (__bf16)aq[j][r];
      k[(qkbase + t) * DD + j * 16 + n] = (__bf16)ak[j][r];
    }
    v8bf pk;  // V transposed [B,H,D,T]: lane writes 8 consecutive t per d
#pragma unroll
    for (int r = 0; r < 8; ++r) pk[r] = (__bf16)av[j][r];
    *(v8bf*)(vt + ((size_t)(b * HH + h) * DD + j * 16 + n) * TT + t0 + hi * 8) = pk;
  }
}

// ---------------- Kernel 2: causal flash attention ----------------
// grid (T/128, B*H), 8 waves/block. K/V tiles (32 keys) are staged into LDS
// once per block (async DMA), triple-buffered with 1 barrier per tile.
__global__ void __launch_bounds__(256) mha_attn(
    const __bf16* __restrict__ qg,  // [B,H,T,D] pre-scaled by D^-0.5*log2(e)
    const __bf16* __restrict__ kg,  // [B,H,T,D]
    const __bf16* __restrict__ vg,  // [B,H,D,T]
    float* __restrict__ out) {      // [B,T,H*D]
  __shared__ __bf16 kbuf[3][32][DD];  // 3 x 4 KB
  __shared__ __bf16 vbuf[3][DD][32];  // 3 x 4 KB
  __shared__ __bf16 plds[8][16][32];  // per-wave P transpose staging, 8 KB

  const int tid = threadIdx.x;
  const int lane = tid & 31;
  const int wave = tid >> 5;
  const int q0 = blockIdx.x * 128 + wave * 16;
  const int b = blockIdx.y / HH;
  const int h = blockIdx.y % HH;

  const __bf16* qb = qg + (size_t)(b * HH + h) * TT * DD;
  const __bf16* kb = kg + (size_t)(b * HH + h) * TT * DD;
  const __bf16* vb = vg + (size_t)(b * HH + h) * DD * TT;

  // cooperative stage of key-tile `it` into buffer `bi`: 2 x b128 per thread
  auto stage_tile = [&](int it, int bi) {
    const int s0 = it * 32;
    const __bf16* gk = kb + (size_t)(s0 + (tid >> 3)) * DD + (tid & 7) * 8;
    stage16(gk, &kbuf[bi][tid >> 3][(tid & 7) * 8]);
    const __bf16* gv = vb + (size_t)(tid >> 2) * TT + s0 + (tid & 3) * 8;
    stage16(gv, &vbuf[bi][tid >> 2][(tid & 3) * 8]);
  };

  v16bf aq0 = load_a(qb + (size_t)q0 * DD, DD, 0, lane);
  v16bf aq1 = load_a(qb + (size_t)q0 * DD, DD, 32, lane);

  v8f o[4];
  float mrow[8], lrow[8];
#pragma unroll
  for (int j = 0; j < 4; ++j) o[j] = zero8();
#pragma unroll
  for (int r = 0; r < 8; ++r) { mrow[r] = -__builtin_inff(); lrow[r] = 0.0f; }

  const int n = lane & 15;
  const int hi = lane >> 4;
  const int q_hi = q0 + 15;
  const int ntiles = blockIdx.x * 4 + 4;  // block-uniform causal bound

  stage_tile(0, 0);

  for (int it = 0; it < ntiles; ++it) {
    const int bi = it % 3;
    if (it + 1 < ntiles) {
      stage_tile(it + 1, (it + 1) % 3);
      asm volatile("s_wait_asynccnt 0x2" ::: "memory");  // tile `it` resident
    } else {
      asm volatile("s_wait_asynccnt 0x0" ::: "memory");
    }
    __syncthreads();  // all waves' shares of tile `it` visible

    const int s0 = it * 32;
    if (s0 <= q_hi) {  // wave-uniform causal predicate
      // S = Q K^T from LDS-resident K tile
      v8f s[2];
#pragma unroll
      for (int g = 0; g < 2; ++g) {
        v8f acc = zero8();
        v16bf bk0 = load_bt(&kbuf[bi][g * 16][0], DD, 0, lane);
        v16bf bk1 = load_bt(&kbuf[bi][g * 16][0], DD, 32, lane);
        acc = wmma_bf16(aq0, bk0, acc);
        acc = wmma_bf16(aq1, bk1, acc);
        s[g] = acc;
      }
      // causal mask on C-layout element (row=q0+r+8*hi, key=s0+16g+n)
#pragma unroll
      for (int g = 0; g < 2; ++g)
#pragma unroll
        for (int r = 0; r < 8; ++r)
          if (s0 + g * 16 + n > q0 + r + hi * 8) s[g][r] = -__builtin_inff();

      // online softmax (log2 domain); rows live across one 16-lane half
#pragma unroll
      for (int r = 0; r < 8; ++r) {
        float v = fmaxf(s[0][r], s[1][r]);
        v = fmaxf(v, __shfl_xor(v, 1));
        v = fmaxf(v, __shfl_xor(v, 2));
        v = fmaxf(v, __shfl_xor(v, 4));
        v = fmaxf(v, __shfl_xor(v, 8));
        float mnew = fmaxf(mrow[r], v);
        float alpha = __builtin_amdgcn_exp2f(mrow[r] - mnew);
        mrow[r] = mnew;
        float p0 = __builtin_amdgcn_exp2f(s[0][r] - mnew);
        float p1 = __builtin_amdgcn_exp2f(s[1][r] - mnew);
        s[0][r] = p0;
        s[1][r] = p1;
        float ps = p0 + p1;
        ps += __shfl_xor(ps, 1);
        ps += __shfl_xor(ps, 2);
        ps += __shfl_xor(ps, 4);
        ps += __shfl_xor(ps, 8);
        lrow[r] = lrow[r] * alpha + ps;
#pragma unroll
        for (int j = 0; j < 4; ++j) o[j][r] *= alpha;
      }

      // transpose P (C-layout -> A-layout) through per-wave LDS
#pragma unroll
      for (int g = 0; g < 2; ++g)
#pragma unroll
        for (int r = 0; r < 8; ++r)
          plds[wave][r + hi * 8][g * 16 + n] = (__bf16)s[g][r];
      asm volatile("s_wait_dscnt 0x0" ::: "memory");  // same-wave DS in-order
      {
        const int prow = lane & 15;
        const int pk = (lane < 16) ? 0 : 8;
        v8bf plo = *(const v8bf*)&plds[wave][prow][pk];
        v8bf phi = *(const v8bf*)&plds[wave][prow][16 + pk];
        v16bf pA = cat8(plo, phi);
#pragma unroll
        for (int j = 0; j < 4; ++j) {
          v16bf bv = load_bt(&vbuf[bi][j * 16][0], 32, 0, lane);
          o[j] = wmma_bf16(pA, bv, o[j]);
        }
      }
    }
  }

  // normalize and write out [B,T,H*D]; coalesced per VGPR row
#pragma unroll
  for (int r = 0; r < 8; ++r) {
    float inv = 1.0f / lrow[r];
    int t = q0 + r + hi * 8;
    float* op = out + ((size_t)b * TT + t) * (HH * DD) + h * DD;
#pragma unroll
    for (int j = 0; j < 4; ++j) op[j * 16 + n] = o[j][r] * inv;
  }
}

extern "C" void kernel_launch(void* const* d_in, const int* in_sizes, int n_in,
                              void* d_out, int out_size, void* d_ws, size_t ws_size,
                              hipStream_t stream) {
  (void)in_sizes; (void)n_in; (void)out_size; (void)ws_size;
  const float* x = (const float*)d_in[0];
  const float* Wq = (const float*)d_in[1];
  const float* Wk = (const float*)d_in[2];
  const float* Wv = (const float*)d_in[3];
  float* out = (float*)d_out;

  const size_t nx = (size_t)BB * TT * CC;
  const size_t nw = (size_t)HH * CC * DD;
  const size_t nq = (size_t)BB * HH * TT * DD;

  __bf16* xb = (__bf16*)d_ws;  // large power-of-two sizes -> aligned
  __bf16* wqt = xb + nx;
  __bf16* wkt = wqt + nw;
  __bf16* wvt = wkt + nw;
  __bf16* qb = wvt + nw;
  __bf16* kbf = qb + nq;
  __bf16* vtb = kbf + nq;  // ~73 MB total workspace

  const float scale_q = 0.125f * 1.44269504088896340736f;  // D^-0.5 * log2(e)

  mha_cvt_x<<<(int)(nx / 4 / 256), 256, 0, stream>>>(x, xb, (int)nx);
  mha_cvt_w<<<(int)(nw / 256), 256, 0, stream>>>(Wq, wqt, scale_q);
  mha_cvt_w<<<(int)(nw / 256), 256, 0, stream>>>(Wk, wkt, 1.0f);
  mha_cvt_w<<<(int)(nw / 256), 256, 0, stream>>>(Wv, wvt, 1.0f);

  mha_proj<<<dim3(TT / 128, BB * HH), 256, 0, stream>>>(xb, wqt, wkt, wvt, qb,
                                                        kbf, vtb);
  mha_attn<<<dim3(TT / 128, BB * HH), 256, 0, stream>>>(qb, kbf, vtb, out);
}
